// MultiheadCrossAttention_49074296324609
// MI455X (gfx1250) — matrix operand order repaired and checked by reading, not compiled, not run on hardware
//
#include <hip/hip_runtime.h>
#include <hip/hip_bf16.h>

typedef __bf16 bf16;
typedef __attribute__((ext_vector_type(16))) __bf16 v16bf;
typedef __attribute__((ext_vector_type(8)))  __bf16 v8bf;
typedef __attribute__((ext_vector_type(8)))  float  v8f;

// ---------------------------------------------------------------- helpers
static __device__ inline v16bf cat8(v8bf lo, v8bf hi) {
  v16bf r;
#pragma unroll
  for (int i = 0; i < 8; ++i) { r[i] = lo[i]; r[i + 8] = hi[i]; }
  return r;
}

// A-fragment (16x32 bf16) loader for row-major A.
// Contiguous K order: lanes 0-15 hold M=lane, K in {0..15}; lanes 16-31 K in {16..31}.
static __device__ inline v16bf load_afrag(const bf16* rowk0, int half) {
  v8bf lo = *(const v8bf*)(rowk0 + 16 * half);
  v8bf hi = *(const v8bf*)(rowk0 + 16 * half + 8);
  return cat8(lo, hi);
}

static __device__ inline float red16max(float v) {
  v = fmaxf(v, __shfl_xor(v, 1, 32));
  v = fmaxf(v, __shfl_xor(v, 2, 32));
  v = fmaxf(v, __shfl_xor(v, 4, 32));
  v = fmaxf(v, __shfl_xor(v, 8, 32));
  return v;
}
static __device__ inline float red16sum(float v) {
  v += __shfl_xor(v, 1, 32);
  v += __shfl_xor(v, 2, 32);
  v += __shfl_xor(v, 4, 32);
  v += __shfl_xor(v, 8, 32);
  return v;
}

// CDNA5 async global->LDS staging (ASYNCcnt-tracked, overlaps with WMMA).
static __device__ inline void async_copy_b128(bf16* dst_lds, const bf16* src_glb) {
  unsigned ldsoff =
      (unsigned)(unsigned long long)(__attribute__((address_space(3))) bf16*)dst_lds;
  unsigned long long gaddr = (unsigned long long)src_glb;
  asm volatile("global_load_async_to_lds_b128 %0, %1, off"
               :: "v"(ldsoff), "v"(gaddr)
               : "memory");
}
static __device__ inline void wait_async_le1() {
  asm volatile("s_wait_asynccnt 0x1" ::: "memory");
}
static __device__ inline void wait_async_le2() {
  asm volatile("s_wait_asynccnt 0x2" ::: "memory");
}

// ---------------------------------------------------------------- small kernels
__global__ void cvt_f32_bf16(const float* __restrict__ s, bf16* __restrict__ d, int n) {
  int i = blockIdx.x * 256 + threadIdx.x;
  if (i < n) d[i] = (bf16)s[i];
}

// Pack W (KxN row-major f32) into B-fragment order:
// Wp[((kt*N + n)*2 + half)*16 + j] = W[(32*kt + 16*half + j)*N + n]
__global__ void pack_weight(const float* __restrict__ W, bf16* __restrict__ Wp, int K, int N) {
  int idx = blockIdx.x * 256 + threadIdx.x;
  if (idx >= K * N) return;
  int j = idx & 15;
  int half = (idx >> 4) & 1;
  int rest = idx >> 5;
  int n = rest % N;
  int kt = rest / N;
  int k = 32 * kt + 16 * half + j;
  Wp[idx] = (bf16)W[(size_t)k * N + n];
}

// bias[b*N+n] = -1e30 if condition row all-zero else 0
__global__ void mask_kernel(const float* __restrict__ c, float* __restrict__ bias,
                            int BN, int CD) {
  int i = blockIdx.x * 256 + threadIdx.x;
  if (i >= BN) return;
  const float* row = c + (size_t)i * CD;
  int allz = 1;
  for (int k = 0; k < CD; ++k) {
    if (row[k] != 0.0f) { allz = 0; break; }
  }
  bias[i] = allz ? -1e30f : 0.0f;
}

// K projection (f32, (B*N) x 1024) -> per-(b,h) B-frag layout over (d,n):
// Kp[((((b*16+h)*2+dt)*N + n)*2 + half)*16 + j] = Kf[(b*N+n)*1024 + (32*dt+16*half+j)*16 + h]
__global__ void pack_k(const float* __restrict__ Kf, bf16* __restrict__ Kp,
                       int B, int Ncond) {
  int idx = blockIdx.x * 256 + threadIdx.x;
  int total = B * 16 * 2 * Ncond * 32;
  if (idx >= total) return;
  int j = idx & 15;
  int half = (idx >> 4) & 1;
  int rest = idx >> 5;
  int n = rest % Ncond;  rest /= Ncond;
  int dt = rest & 1;     rest >>= 1;
  int h = rest & 15;
  int b = rest >> 4;
  int d = 32 * dt + 16 * half + j;
  Kp[idx] = (bf16)Kf[((size_t)(b * Ncond + n)) * 1024 + d * 16 + h];
}

// V projection -> per-(b,h) B-frag layout over (n,d):
// Vp[((((b*16+h)*NT + nt)*64 + d)*2 + half)*16 + j] = Vf[(b*N + 32*nt + 16*half + j)*1024 + d*16 + h]
__global__ void pack_v(const float* __restrict__ Vf, bf16* __restrict__ Vp,
                       int B, int Ncond) {
  int idx = blockIdx.x * 256 + threadIdx.x;
  int NT = Ncond / 32;
  int total = B * 16 * NT * 64 * 32;
  if (idx >= total) return;
  int j = idx & 15;
  int half = (idx >> 4) & 1;
  int rest = idx >> 5;
  int d = rest & 63;   rest >>= 6;
  int nt = rest % NT;  rest /= NT;
  int h = rest & 15;
  int b = rest >> 4;
  int n = 32 * nt + 16 * half + j;
  Vp[idx] = (bf16)Vf[((size_t)(b * Ncond + n)) * 1024 + d * 16 + h];
}

// ---------------------------------------------------------------- GEMM (bf16 -> f32 acc)
// Block = 256 thr = 8 waves; wave w owns rows [blk.x*128 + w*16), cols [blk.y*64, +64).
// Packed-B tile per (kt, n-block) is one contiguous 4KB run -> double-buffered
// async global->LDS staging (256 threads x 16B), WMMA overlaps next tile's DMA.
// Branchless: last iteration re-stages the final tile into the dead buffer.
__global__ __launch_bounds__(256)
void gemm_bf16(const bf16* __restrict__ A, const bf16* __restrict__ Bp,
               float* __restrict__ Cf, bf16* __restrict__ Cb,
               int M, int N, int K) {
  __shared__ bf16 bstage[2][2048];   // 2 x 4KB B tiles (64 cols x 32 K)
  const int tid = threadIdx.x;
  const int lane = tid & 31;
  const int wid  = tid >> 5;
  const int lanelo = lane & 15;
  const int half = lane >> 4;
  const int m0 = blockIdx.x * 128 + wid * 16;
  const int n0 = blockIdx.y * 64;
  const int KT = K >> 5;

  v8f acc[4] = {};
  const bf16* Arow = A + (size_t)(m0 + lanelo) * K;
  const bf16* Btile = Bp + ((size_t)n0) * 32 + tid * 8;  // +kt*N*32 per step

  // prologue: stage kt=0 into buf0
  async_copy_b128(&bstage[0][tid * 8], Btile);

  for (int kt = 0; kt < KT; ++kt) {
    const int buf = kt & 1;
    const int ktn = (kt + 1 < KT) ? kt + 1 : kt;     // clamped (dummy re-stage at end)
    async_copy_b128(&bstage[buf ^ 1][tid * 8], Btile + (size_t)ktn * N * 32);
    wait_async_le1();     // retire stage(kt); stage(kt+1) stays in flight
    __syncthreads();      // all waves staged tile kt

    v16bf a = load_afrag(Arow, half);
    Arow += 32;
    __builtin_prefetch(Arow, 0, 1);   // speculative; dropped if invalid

    const bf16* bb = &bstage[buf][0];
#pragma unroll
    for (int t = 0; t < 4; ++t) {
      v16bf b = *(const v16bf*)(bb + (t * 16 + lanelo) * 32 + half * 16);
      acc[t] = __builtin_amdgcn_wmma_f32_16x16x32_bf16(
          false, a, false, b, (short)0, acc[t], false, false);
    }
    __syncthreads();      // reads of tile kt done before its buffer is re-staged
  }

#pragma unroll
  for (int t = 0; t < 4; ++t) {
#pragma unroll
    for (int r = 0; r < 8; ++r) {
      int row = m0 + 8 * half + r;
      int col = n0 + t * 16 + lanelo;
      if (Cf) Cf[(size_t)row * N + col] = acc[t][r];
      else    Cb[(size_t)row * N + col] = (bf16)acc[t][r];
    }
  }
}

// ---------------------------------------------------------------- flash attention
// Grid: (S/128, B*H). Block = 256 thr = 8 waves; wave w handles 16 query rows.
// K/V for each 32-key chunk (8KB: 2x2KB K runs + 4KB V run, contiguous in packed
// layouts) are double-buffer async-staged into LDS and shared by all 8 waves.
__global__ __launch_bounds__(256)
void attn_kernel(const bf16* __restrict__ Qb, const bf16* __restrict__ Kp,
                 const bf16* __restrict__ Vp, const float* __restrict__ bias,
                 bf16* __restrict__ Ob, int S, int Ncond) {
  __shared__ bf16 kvstage[2][4096];  // [buf][ Kdt0:0..1023 | Kdt1:1024..2047 | V:2048..4095 ]
  __shared__ bf16 plds[8][16 * 32];
  const int tid = threadIdx.x;
  const int lane = tid & 31;
  const int wv   = tid >> 5;
  const int lanelo = lane & 15;
  const int half = lane >> 4;
  const int bh = blockIdx.y;          // b*16 + h
  const int b = bh >> 4, h = bh & 15;
  const int s0 = blockIdx.x * 128 + wv * 16;
  const float scale = 0.125f;         // 64^-0.5
  const int NT = Ncond / 32;

  const bf16* KpBH = Kp + (size_t)bh * 2 * Ncond * 32;
  const bf16* VpBH = Vp + (size_t)bh * NT * 64 * 32;

  // per-thread staging sources: 512 chunks of 16B per tile, 2 per thread
  const size_t koff = (tid < 128)
      ? ((size_t)tid * 8)                              // K dt0 run
      : ((size_t)Ncond * 32 + (size_t)(tid - 128) * 8);// K dt1 run
  // stage(nt, buf): K chunk + V chunk for this thread
  auto stage = [&](int nt, int buf) {
    async_copy_b128(&kvstage[buf][tid * 8],        KpBH + (size_t)nt * 1024 + koff);
    async_copy_b128(&kvstage[buf][2048 + tid * 8], VpBH + (size_t)nt * 2048 + tid * 8);
  };

  // Q fragments, K-dim = 64 -> two 16x32 A-frags
  const bf16* qrow = Qb + ((size_t)(b * S + s0 + lanelo)) * 1024 + h * 64;
  v16bf qf[2];
  qf[0] = load_afrag(qrow, half);
  qf[1] = load_afrag(qrow + 32, half);

  v8f o[4] = {};
  float m[8], l[8];
#pragma unroll
  for (int r = 0; r < 8; ++r) { m[r] = -1e30f; l[r] = 0.0f; }

  bf16* myp = &plds[wv][0];

  stage(0, 0);   // prologue

  for (int nt = 0; nt < NT; ++nt) {
    const int buf = nt & 1;
    const int ntn = (nt + 1 < NT) ? nt + 1 : nt;     // clamped dummy re-stage at end
    stage(ntn, buf ^ 1);
    wait_async_le2();     // retire stage(nt); next stage stays in flight
    __syncthreads();      // tile nt published to all waves

    const bf16* kv = &kvstage[buf][0];

    // ---- scores: two 16x16 tiles covering 32 key columns
    v8f sc[2] = {};
#pragma unroll
    for (int u = 0; u < 2; ++u) {
#pragma unroll
      for (int dt = 0; dt < 2; ++dt) {
        v16bf kf = *(const v16bf*)(kv + dt * 1024 + (u * 16 + lanelo) * 32 + half * 16);
        sc[u] = __builtin_amdgcn_wmma_f32_16x16x32_bf16(
            false, qf[dt], false, kf, (short)0, sc[u], false, false);
      }
    }
    float bias0 = bias[b * Ncond + nt * 32 + lanelo];
    float bias1 = bias[b * Ncond + nt * 32 + 16 + lanelo];

    // ---- online softmax update (per-row stats replicated across 16-lane groups)
#pragma unroll
    for (int r = 0; r < 8; ++r) {
      float sv0 = sc[0][r] * scale + bias0;
      float sv1 = sc[1][r] * scale + bias1;
      float mo = m[r];
      float mn = fmaxf(mo, red16max(fmaxf(sv0, sv1)));
      float cr = __expf(mo - mn);
      float p0 = __expf(sv0 - mn);
      float p1 = __expf(sv1 - mn);
      l[r] = l[r] * cr + red16sum(p0 + p1);
      m[r] = mn;
      int row = 8 * half + r;           // C-layout row index within tile
      myp[row * 32 + lanelo]      = (bf16)p0;
      myp[row * 32 + 16 + lanelo] = (bf16)p1;
#pragma unroll
      for (int t = 0; t < 4; ++t) o[t][r] *= cr;
    }

    // ---- reshape P (C-layout) -> A-fragment via per-wave LDS (DS ops in-order per wave)
    v8bf plo = *(const v8bf*)(myp + lanelo * 32 + 16 * half);
    v8bf phi = *(const v8bf*)(myp + lanelo * 32 + 16 * half + 8);
    v16bf pf = cat8(plo, phi);

    // ---- O += P @ V  (four 16-wide d tiles)
#pragma unroll
    for (int t = 0; t < 4; ++t) {
      v16bf vf = *(const v16bf*)(kv + 2048 + (t * 16 + lanelo) * 32 + half * 16);
      o[t] = __builtin_amdgcn_wmma_f32_16x16x32_bf16(
          false, pf, false, vf, (short)0, o[t], false, false);
    }
    __syncthreads();      // K/V reads of tile nt done before buffer re-stage
  }

  // ---- epilogue: normalize and store bf16 (feeds final GEMM)
#pragma unroll
  for (int r = 0; r < 8; ++r) {
    float inv = 1.0f / l[r];
    int row = s0 + 8 * half + r;
#pragma unroll
    for (int t = 0; t < 4; ++t) {
      Ob[((size_t)(b * S + row)) * 1024 + h * 64 + t * 16 + lanelo] =
          (bf16)(o[t][r] * inv);
    }
  }
}

// ---------------------------------------------------------------- launch
extern "C" void kernel_launch(void* const* d_in, const int* in_sizes, int n_in,
                              void* d_out, int out_size, void* d_ws, size_t ws_size,
                              hipStream_t stream) {
  const float* x  = (const float*)d_in[0];
  const float* c  = (const float*)d_in[1];
  const float* Wq = (const float*)d_in[2];
  const float* Wk = (const float*)d_in[3];
  const float* Wv = (const float*)d_in[4];
  const float* Wo = (const float*)d_in[5];
  float* out = (float*)d_out;

  const int B = 2, S = 4096, CH = 1024, N = 1024, H = 16;
  const int MS = B * S;   // 8192 query rows
  const int MN = B * N;   // 2048 condition rows

  char* ws = (char*)d_ws;
  size_t off = 0;
  auto alloc = [&](size_t bytes) -> void* {
    void* p = ws + off;
    off += (bytes + 255) & ~(size_t)255;
    return p;
  };
  bf16* xb  = (bf16*)alloc((size_t)MS * CH * 2);
  bf16* cb  = (bf16*)alloc((size_t)MN * CH * 2);
  bf16* Wqp = (bf16*)alloc((size_t)CH * CH * 2);
  bf16* Wkp = (bf16*)alloc((size_t)CH * CH * 2);
  bf16* Wvp = (bf16*)alloc((size_t)CH * CH * 2);
  bf16* Wop = (bf16*)alloc((size_t)CH * CH * 2);
  bf16* Qb  = (bf16*)alloc((size_t)MS * CH * 2);
  float* scratch = (float*)alloc((size_t)MN * CH * 4);
  bf16* Kpk = (bf16*)alloc((size_t)B * H * 64 * N * 2);
  bf16* Vpk = (bf16*)alloc((size_t)B * H * N * 64 * 2);
  bf16* Ob  = (bf16*)alloc((size_t)MS * CH * 2);
  float* bias = (float*)alloc((size_t)B * N * 4);

  dim3 blk256(256);
  cvt_f32_bf16<<<(MS * CH + 255) / 256, blk256, 0, stream>>>(x, xb, MS * CH);
  cvt_f32_bf16<<<(MN * CH + 255) / 256, blk256, 0, stream>>>(c, cb, MN * CH);
  pack_weight<<<(CH * CH + 255) / 256, blk256, 0, stream>>>(Wq, Wqp, CH, CH);
  pack_weight<<<(CH * CH + 255) / 256, blk256, 0, stream>>>(Wk, Wkp, CH, CH);
  pack_weight<<<(CH * CH + 255) / 256, blk256, 0, stream>>>(Wv, Wvp, CH, CH);
  pack_weight<<<(CH * CH + 255) / 256, blk256, 0, stream>>>(Wo, Wop, CH, CH);
  mask_kernel<<<(B * N + 255) / 256, blk256, 0, stream>>>(c, bias, B * N, CH);

  dim3 gQ(MS / 128, CH / 64), gC(MN / 128, CH / 64);
  // Q = x @ Wq  -> bf16 directly
  gemm_bf16<<<gQ, blk256, 0, stream>>>(xb, Wqp, nullptr, Qb, MS, CH, CH);
  // K = c @ Wk -> f32 scratch -> packed bf16
  gemm_bf16<<<gC, blk256, 0, stream>>>(cb, Wkp, scratch, nullptr, MN, CH, CH);
  {
    int tot = B * H * 2 * N * 32;
    pack_k<<<(tot + 255) / 256, blk256, 0, stream>>>(scratch, Kpk, B, N);
  }
  // V = c @ Wv -> f32 scratch -> packed bf16
  gemm_bf16<<<gC, blk256, 0, stream>>>(cb, Wvp, scratch, nullptr, MN, CH, CH);
  {
    int tot = B * H * (N / 32) * 64 * 32;
    pack_v<<<(tot + 255) / 256, blk256, 0, stream>>>(scratch, Vpk, B, N);
  }
  // attention
  attn_kernel<<<dim3(S / 128, B * H), blk256, 0, stream>>>(Qb, Kpk, Vpk, bias, Ob, S, N);
  // out = attn_out @ Wo (f32 output)
  gemm_bf16<<<gQ, blk256, 0, stream>>>(Ob, Wop, out, nullptr, MS, CH, CH);
}
